// OutputBlock_22737556865508
// MI455X (gfx1250) — matrix operand order repaired and checked by reading, not compiled, not run on hardware
//
#include <hip/hip_runtime.h>
#include <hip/hip_bf16.h>

typedef __bf16 v16bf __attribute__((ext_vector_type(16)));
typedef float  v8f   __attribute__((ext_vector_type(8)));

#define EMB  128
#define NRBF 6

// ---------------- CSR build ----------------

__global__ void kzero_u32(unsigned* __restrict__ p, int n) {
  int i = blockIdx.x * blockDim.x + threadIdx.x;
  if (i < n) p[i] = 0u;
}

__global__ void khist(const long long* __restrict__ nbr, unsigned* __restrict__ counts, int E) {
  int e = blockIdx.x * blockDim.x + threadIdx.x;
  if (e < E) {
    int dst = (int)nbr[2 * e + 1];
    atomicAdd(&counts[dst], 1u);
  }
}

// single-workgroup Hillis-Steele scan over N counts -> offsets[N+1], cursor[N]=start
__global__ __launch_bounds__(1024) void kscan(const unsigned* __restrict__ counts,
                                              unsigned* __restrict__ offsets,
                                              unsigned* __restrict__ cursor, int n) {
  __shared__ unsigned s[1024];
  __shared__ unsigned carry_s;
  int tid = threadIdx.x;
  if (tid == 0) { carry_s = 0u; offsets[0] = 0u; }
  __syncthreads();
  for (int base = 0; base < n; base += 1024) {
    unsigned v = (base + tid < n) ? counts[base + tid] : 0u;
    s[tid] = v;
    __syncthreads();
    for (int off = 1; off < 1024; off <<= 1) {
      unsigned t = (tid >= off) ? s[tid - off] : 0u;
      __syncthreads();
      s[tid] += t;
      __syncthreads();
    }
    unsigned incl  = s[tid];
    unsigned carry = carry_s;
    if (base + tid < n) {
      offsets[base + tid + 1] = carry + incl;
      cursor[base + tid]      = carry + incl - v;  // exclusive prefix = bucket start
    }
    __syncthreads();
    if (tid == 1023) carry_s = carry + incl;
    __syncthreads();
  }
}

__global__ void kscatter(const long long* __restrict__ nbr, unsigned* __restrict__ cursor,
                         unsigned* __restrict__ eids, int E) {
  int e = blockIdx.x * blockDim.x + threadIdx.x;
  if (e < E) {
    int dst = (int)nbr[2 * e + 1];
    unsigned p = atomicAdd(&cursor[dst], 1u);
    eids[p] = (unsigned)e;
  }
}

// ------------- edge transform + gather-reduce (bandwidth bound: 307 MB of m_ji) -------------
// one wave per atom; lane owns 4 channels (32 lanes * 4 = 128)
__global__ __launch_bounds__(128) void kgather(const float* __restrict__ m_ji,
                                               const float* __restrict__ e_rbf,
                                               const float* __restrict__ W_edge,
                                               const unsigned* __restrict__ offsets,
                                               const unsigned* __restrict__ eids,
                                               float* __restrict__ node, int N) {
  int lane = threadIdx.x & 31;
  int atom = blockIdx.x * 4 + (threadIdx.x >> 5);
  if (atom >= N) return;

  float we[NRBF][4];
#pragma unroll
  for (int k = 0; k < NRBF; ++k)
#pragma unroll
    for (int j = 0; j < 4; ++j) we[k][j] = W_edge[k * EMB + lane * 4 + j];

  float a0 = 0.f, a1 = 0.f, a2 = 0.f, a3 = 0.f;
  unsigned i0 = offsets[atom], i1 = offsets[atom + 1];
  for (unsigned i = i0; i < i1; ++i) {
    unsigned eid = eids[i];
    if (i + 1 < i1) {
      unsigned en = eids[i + 1];
      __builtin_prefetch(m_ji + (size_t)en * EMB + lane * 4, 0, 1);  // global_prefetch_b8
    }
    const float* r = e_rbf + (size_t)eid * NRBF;
    float r0 = r[0], r1 = r[1], r2 = r[2], r3 = r[3], r4 = r[4], r5 = r[5];
    const float4 m = *((const float4*)(m_ji + (size_t)eid * EMB) + lane);
    float c0 = r0*we[0][0] + r1*we[1][0] + r2*we[2][0] + r3*we[3][0] + r4*we[4][0] + r5*we[5][0];
    float c1 = r0*we[0][1] + r1*we[1][1] + r2*we[2][1] + r3*we[3][1] + r4*we[4][1] + r5*we[5][1];
    float c2 = r0*we[0][2] + r1*we[1][2] + r2*we[2][2] + r3*we[3][2] + r4*we[4][2] + r5*we[5][2];
    float c3 = r0*we[0][3] + r1*we[1][3] + r2*we[2][3] + r3*we[3][3] + r4*we[4][3] + r5*we[5][3];
    a0 += c0 * m.x; a1 += c1 * m.y; a2 += c2 * m.z; a3 += c3 * m.w;
  }
  float4 outv = make_float4(a0, a1, a2, a3);
  *((float4*)(node + (size_t)atom * EMB) + lane) = outv;
}

// ------------- weight repack: f32 row-major -> bf16 WMMA B fragments -------------
// layout: [(l*8+nt)*4+kc][lane][16 elems]; B 32x16: lane -> N=lane%16, h=lane/16; elem 2v+j -> K=32kc+16h+2v+j
__global__ void kpack(const float* __restrict__ W1, const float* __restrict__ W2,
                      const float* __restrict__ W3, const float* __restrict__ W4,
                      __bf16* __restrict__ wp) {
  int t = blockIdx.x * blockDim.x + threadIdx.x;
  if (t >= 4 * EMB * EMB) return;
  int e    = t & 15;
  int lane = (t >> 4) & 31;
  int kc   = (t >> 9) & 3;
  int nt   = (t >> 11) & 7;
  int l    = (t >> 14) & 3;
  int v = e >> 1, j = e & 1;
  int Nn = lane & 15, h = lane >> 4;
  int K = kc * 32 + h * 16 + v * 2 + j;
  int n = nt * 16 + Nn;
  const float* W = (l == 0) ? W1 : (l == 1) ? W2 : (l == 2) ? W3 : W4;
  wp[t] = (__bf16)W[K * EMB + n];
}

// ------------- fused 4-layer MLP with swish, bf16 WMMA, f32 accumulate -------------
// 4 waves/block, one 16x128 row tile per wave; LDS staging does the D->A transpose between layers.
__global__ __launch_bounds__(128) void kmlp(const float* __restrict__ node,
                                            const v16bf* __restrict__ wp,
                                            const float* __restrict__ b1,
                                            const float* __restrict__ b2,
                                            const float* __restrict__ b3,
                                            float* __restrict__ out, int N) {
  __shared__ __attribute__((aligned(16))) float stg[4][16][132];  // stride 132 dw -> conflict-free
  int lane = threadIdx.x & 31;
  int w    = threadIdx.x >> 5;
  int tilesTotal = (N + 15) >> 4;
  int tileIdx = blockIdx.x * 4 + w;
  bool wr  = tileIdx < tilesTotal;
  int tile = wr ? tileIdx : 0;  // invalid waves compute on tile 0, skip store (barrier-safe)
  int rbase = tile * 16;

  // coalesced load of the 16x128 f32 tile into staging
#pragma unroll
  for (int r = 0; r < 16; ++r) {
    int row = rbase + r;
    float4 val = (row < N) ? *((const float4*)(node + (size_t)row * EMB) + lane)
                           : make_float4(0.f, 0.f, 0.f, 0.f);
    *(float4*)&stg[w][r][lane * 4] = val;
  }
  __syncthreads();

  int ln = lane & 15, h = lane >> 4;  // A row M=ln; C/D column N=ln
  v16bf a[4];
  for (int l = 0; l < 4; ++l) {
    // build A fragments (contiguous k order, same as kpack's B): elem 2v+j -> K = 32kc + 16h + 2v + j
#pragma unroll
    for (int kc = 0; kc < 4; ++kc) {
#pragma unroll
      for (int v = 0; v < 8; ++v) {
        int vp = v & 3, g = v >> 2;
        int K0 = kc * 32 + h * 16 + v * 2;
        a[kc][2 * v]     = (__bf16)stg[w][ln][K0];
        a[kc][2 * v + 1] = (__bf16)stg[w][ln][K0 + 1];
      }
    }
    __syncthreads();

    const float* bptr = (l == 0) ? b1 : (l == 1) ? b2 : b3;
#pragma unroll
    for (int nt = 0; nt < 8; ++nt) {
      float bv = (l < 3) ? bptr[nt * 16 + ln] : 0.f;  // bias depends only on column N
      v8f acc;
#pragma unroll
      for (int v = 0; v < 8; ++v) acc[v] = bv;
#pragma unroll
      for (int kc = 0; kc < 4; ++kc) {
        v16bf bfrag = wp[((((l * 8 + nt) * 4) + kc) * 32) + lane];  // coalesced 32B/lane
        acc = __builtin_amdgcn_wmma_f32_16x16x32_bf16(
            false, a[kc], false, bfrag, (short)0, acc, false, false);
      }
      if (l < 3) {
        // swish: x * sigmoid(x); v_rcp_f32 instead of IEEE divide (avoids
        // v_div_scale/v_div_fixup expansion — ~10x fewer VALU ops here)
#pragma unroll
        for (int v = 0; v < 8; ++v) {
          float x = acc[v];
          acc[v] = x * __builtin_amdgcn_rcpf(1.f + __expf(-x));
        }
      }
      // D (f32 C/D layout): (M = v + 8h, N = ln) back to staging for next layer / store
#pragma unroll
      for (int v = 0; v < 8; ++v) stg[w][v + 8 * h][nt * 16 + ln] = acc[v];
    }
    __syncthreads();
  }

  if (wr) {
#pragma unroll
    for (int r = 0; r < 16; ++r) {
      int row = rbase + r;
      if (row < N) *((float4*)(out + (size_t)row * EMB) + lane) = *(float4*)&stg[w][r][lane * 4];
    }
  }
}

// ---------------- launch ----------------

extern "C" void kernel_launch(void* const* d_in, const int* in_sizes, int n_in,
                              void* d_out, int out_size, void* d_ws, size_t ws_size,
                              hipStream_t stream) {
  const float*     m_ji   = (const float*)d_in[0];
  const float*     e_rbf  = (const float*)d_in[1];
  const long long* nbr    = (const long long*)d_in[2];
  const float*     W_edge = (const float*)d_in[4];
  const float*     W1     = (const float*)d_in[5];
  const float*     b1     = (const float*)d_in[6];
  const float*     W2     = (const float*)d_in[7];
  const float*     b2     = (const float*)d_in[8];
  const float*     W3     = (const float*)d_in[9];
  const float*     b3     = (const float*)d_in[10];
  const float*     W4     = (const float*)d_in[11];
  float*           out    = (float*)d_out;

  int E = in_sizes[0] / EMB;   // 600000
  int N = out_size    / EMB;   // 20000

  char*  ws  = (char*)d_ws;
  size_t off = 0;
  auto alloc = [&](size_t bytes) -> char* {
    char* p = ws + off;
    off = (off + bytes + 255) & ~(size_t)255;
    return p;
  };
  unsigned* counts  = (unsigned*)alloc((size_t)N * 4);
  unsigned* offsets = (unsigned*)alloc((size_t)(N + 1) * 4);
  unsigned* cursor  = (unsigned*)alloc((size_t)N * 4);
  unsigned* eids    = (unsigned*)alloc((size_t)E * 4);
  float*    node    = (float*)alloc((size_t)N * EMB * 4);
  __bf16*   wp      = (__bf16*)alloc((size_t)4 * EMB * EMB * 2);
  (void)ws_size; (void)n_in;

  kzero_u32<<<(N + 255) / 256, 256, 0, stream>>>(counts, N);
  khist<<<(E + 255) / 256, 256, 0, stream>>>(nbr, counts, E);
  kscan<<<1, 1024, 0, stream>>>(counts, offsets, cursor, N);
  kscatter<<<(E + 255) / 256, 256, 0, stream>>>(nbr, cursor, eids, E);
  kgather<<<(N + 3) / 4, 128, 0, stream>>>(m_ji, e_rbf, W_edge, offsets, eids, node, N);
  kpack<<<(4 * EMB * EMB + 255) / 256, 256, 0, stream>>>(W1, W2, W3, W4, wp);
  int tiles = (N + 15) / 16;
  kmlp<<<(tiles + 3) / 4, 128, 0, stream>>>(node, (const v16bf*)wp, b1, b2, b3, out, N);
}
